// RoutedExperts_2808908611701
// MI455X (gfx1250) — matrix operand order repaired and checked by reading, not compiled, not run on hardware
//
#include <hip/hip_runtime.h>

// Problem constants (match reference)
#define DMODEL 2048
#define HID    1024
#define NEXP   8

#define LDK        32   // K per WMMA step (bf16)
#define LDS_STRIDE 40   // 32 + 8 pad (bf16 elements) to break LDS bank conflicts
#define TILE       128  // block tile: 128 x 128, 8 waves of wave32

typedef __attribute__((ext_vector_type(16))) __bf16         bf16x16;
typedef __attribute__((ext_vector_type(8)))  unsigned short u16x8;
typedef __attribute__((ext_vector_type(16))) unsigned short u16x16;
typedef __attribute__((ext_vector_type(8)))  float          f32x8;

// Pack two f32 -> one dword of two bf16 (lo = first, hi = second).
// One round-bias add per element + a single v_perm_b32 per pair.
__device__ __forceinline__ unsigned pack_bf16x2(float lo, float hi) {
    unsigned ul = __builtin_bit_cast(unsigned, lo) + 0x8000u;
    unsigned uh = __builtin_bit_cast(unsigned, hi) + 0x8000u;
    // V_PERM_B32: sel bytes 0-3 -> src1, 4-7 -> src0.
    // 0x07060302 -> {src0[31:16], src1[31:16]} = {bf16(hi), bf16(lo)}
    return __builtin_amdgcn_perm(uh, ul, 0x07060302u);
}

__device__ __forceinline__ unsigned short f2bf_raw(float f) {
    unsigned u = __builtin_bit_cast(unsigned, f) + 0x8000u;
    return (unsigned short)(u >> 16);
}

// Stage 16 consecutive f32 -> 16 bf16 in LDS (two uint4 stores).
__device__ __forceinline__ void stage16(const float* __restrict__ src,
                                        unsigned short* __restrict__ dst) {
    const float4* s4 = (const float4*)src;
    float4 a0 = s4[0], a1 = s4[1], a2 = s4[2], a3 = s4[3];
    uint4 w0, w1;
    w0.x = pack_bf16x2(a0.x, a0.y); w0.y = pack_bf16x2(a0.z, a0.w);
    w0.z = pack_bf16x2(a1.x, a1.y); w0.w = pack_bf16x2(a1.z, a1.w);
    w1.x = pack_bf16x2(a2.x, a2.y); w1.y = pack_bf16x2(a2.z, a2.w);
    w1.z = pack_bf16x2(a3.x, a3.y); w1.w = pack_bf16x2(a3.z, a3.w);
    ((uint4*)dst)[0] = w0;
    ((uint4*)dst)[1] = w1;
}

// A fragment: 16x32 bf16, lanes 0-15 = M rows; e[0..7] -> K = half*8+i,
// e[8..15] -> K = 16+half*8+i (half = lane>>4).
__device__ __forceinline__ u16x16 ldsA_frag(const unsigned short* base, int m0, int lane) {
    const int m    = m0 + (lane & 15);
    const int half = lane >> 4;
    const unsigned short* p = base + m * LDS_STRIDE + half * 8;
    u16x8 lo = *(const u16x8*)p;
    u16x8 hi = *(const u16x8*)(p + 16);
    return __builtin_shufflevector(lo, hi, 0,1,2,3,4,5,6,7,8,9,10,11,12,13,14,15);
}

// B fragment: 32x16 (KxN), lane = N col; e[0..7] -> K = (lane>>4)*8 + i, e[8..15] -> K = 16 + (lane>>4)*8 + (i-8).
// LDS B tile stored [N][K] row-major.
__device__ __forceinline__ u16x16 ldsB_frag(const unsigned short* base, int n0, int lane) {
    const int n  = n0 + (lane & 15);
    const int kh = lane >> 4;
    const unsigned short* p = base + n * LDS_STRIDE + kh * 8;
    u16x8 lo = *(const u16x8*)p;
    u16x8 hi = *(const u16x8*)(p + 16);
    return __builtin_shufflevector(lo, hi, 0,1,2,3,4,5,6,7,8,9,10,11,12,13,14,15);
}

__device__ __forceinline__ f32x8 wmma_bf16(u16x16 a, u16x16 b, f32x8 c) {
    return __builtin_amdgcn_wmma_f32_16x16x32_bf16(
        false, __builtin_bit_cast(bf16x16, a),
        false, __builtin_bit_cast(bf16x16, b),
        (short)0, c, false, false);
}

// Which expert owns token-row row0; rows_valid = rows of this TILE inside the expert.
__device__ __forceinline__ int find_expert(const int* __restrict__ bse, int row0, int* rows_valid) {
    int e = 0, off = 0;
    bool found = false;
#pragma unroll
    for (int i = 0; i < NEXP; ++i) {
        int s = bse[i];
        if (!found) {
            if (row0 < off + s) { e = i; found = true; }
            else off += s;
        }
    }
    int rv = off + bse[e] - row0;
    if (rv > TILE) rv = TILE;
    if (rv < 0)    rv = 0;
    *rows_valid = rv;
    return e;
}

// ---------------------------------------------------------------------------
// Kernel 1: fused up/gate GEMM + SiLU*mul -> h (bf16) in workspace.
// Block: 256 threads = 8 waves (2 M x 4 N), tile 128x128, wave tile 64x32.
// Per K-step per wave: 4 A frags x 2 B cols x {gate,up} = 16 WMMAs.
// ---------------------------------------------------------------------------
__global__ __launch_bounds__(256) void moe_upgate_silu_kernel(
    const float* __restrict__ x, const float* __restrict__ wug,
    const int* __restrict__ bse, unsigned short* __restrict__ h, int n_tokens)
{
    __shared__ unsigned short lA [TILE * LDS_STRIDE];
    __shared__ unsigned short lBg[TILE * LDS_STRIDE];
    __shared__ unsigned short lBu[TILE * LDS_STRIDE];

    const int tile_n = blockIdx.x;           // hidden-column tile (HID/TILE)
    const int row0   = blockIdx.y * TILE;    // global token row
    if (row0 >= n_tokens) return;

    int rows_valid;
    const int e = find_expert(bse, row0, &rows_valid);

    const int t    = threadIdx.x;
    const int lane = t & 31;
    const int wave = t >> 5;
    const int wm   = (wave >> 2) * 64;       // wave M offset (0 or 64)
    const int wn   = (wave & 3)  * 32;       // wave N offset (0..96)

    // staging: 2 threads per row, 16 f32 each (128 rows x 32 cols)
    const int srow = t >> 1;
    const int scol = (t & 1) * 16;

    const float* xrow = x + (size_t)(row0 + srow) * DMODEL + scol;
    const float* wg   = wug + (size_t)e * (2 * HID) * DMODEL
                            + (size_t)(tile_n * TILE + srow) * DMODEL + scol;
    const float* wu   = wg + (size_t)HID * DMODEL;

    unsigned short* dA = lA  + srow * LDS_STRIDE + scol;
    unsigned short* dG = lBg + srow * LDS_STRIDE + scol;
    unsigned short* dU = lBu + srow * LDS_STRIDE + scol;

    f32x8 accg[4][2] = {};
    f32x8 accu[4][2] = {};

    for (int k0 = 0; k0 < DMODEL; k0 += LDK) {
        if (k0 + LDK < DMODEL) {
            __builtin_prefetch(xrow + k0 + LDK, 0, 1);
            __builtin_prefetch(wg   + k0 + LDK, 0, 1);
            __builtin_prefetch(wu   + k0 + LDK, 0, 1);
        }
        stage16(xrow + k0, dA);
        stage16(wg   + k0, dG);
        stage16(wu   + k0, dU);
        __syncthreads();
#pragma unroll
        for (int mi = 0; mi < 4; ++mi) {
            u16x16 af = ldsA_frag(lA, wm + mi * 16, lane);
#pragma unroll
            for (int ni = 0; ni < 2; ++ni) {
                u16x16 bg = ldsB_frag(lBg, wn + ni * 16, lane);
                u16x16 bu = ldsB_frag(lBu, wn + ni * 16, lane);
                accg[mi][ni] = wmma_bf16(af, bg, accg[mi][ni]);
                accu[mi][ni] = wmma_bf16(af, bu, accu[mi][ni]);
            }
        }
        __syncthreads();
    }

    // Epilogue: h = silu(g) * u, store bf16
    const int cl = lane & 15;
    const int rh = (lane >> 4) * 8;
#pragma unroll
    for (int mi = 0; mi < 4; ++mi) {
#pragma unroll
        for (int ni = 0; ni < 2; ++ni) {
            f32x8 g = accg[mi][ni];
            f32x8 u = accu[mi][ni];
#pragma unroll
            for (int v = 0; v < 8; ++v) {
                int r = wm + mi * 16 + rh + v;
                if (r < rows_valid) {
                    float gs = g[v];
                    float hv = (gs / (1.0f + __expf(-gs))) * u[v];
                    h[(size_t)(row0 + r) * HID + tile_n * TILE + wn + ni * 16 + cl] =
                        f2bf_raw(hv);
                }
            }
        }
    }
}

// ---------------------------------------------------------------------------
// Kernel 2: down projection: out = h @ Wd_e^T (K = HID). A already bf16.
// Block: 256 threads = 8 waves, tile 128x128, wave tile 64x32, 8 WMMAs/K-step.
// ---------------------------------------------------------------------------
__global__ __launch_bounds__(256) void moe_down_kernel(
    const unsigned short* __restrict__ h, const float* __restrict__ wdown,
    const int* __restrict__ bse, float* __restrict__ out, int n_tokens)
{
    __shared__ unsigned short lA[TILE * LDS_STRIDE];
    __shared__ unsigned short lB[TILE * LDS_STRIDE];

    const int tile_n = blockIdx.x;           // d_model tile (DMODEL/TILE)
    const int row0   = blockIdx.y * TILE;
    if (row0 >= n_tokens) return;

    int rows_valid;
    const int e = find_expert(bse, row0, &rows_valid);

    const int t    = threadIdx.x;
    const int lane = t & 31;
    const int wave = t >> 5;
    const int wm   = (wave >> 2) * 64;
    const int wn   = (wave & 3)  * 32;

    const int srow = t >> 1;
    const int scol = (t & 1) * 16;

    const unsigned short* hrow = h + (size_t)(row0 + srow) * HID + scol;
    // w_down is [DMODEL][NEXP*HID] row-major; B row n uses cols e*HID + k
    const float* wb = wdown + (size_t)(tile_n * TILE + srow) * (NEXP * HID)
                            + (size_t)e * HID + scol;

    unsigned short* dA = lA + srow * LDS_STRIDE + scol;
    unsigned short* dB = lB + srow * LDS_STRIDE + scol;

    f32x8 acc[4][2] = {};

    for (int k0 = 0; k0 < HID; k0 += LDK) {
        if (k0 + LDK < HID) {
            __builtin_prefetch(hrow + k0 + LDK, 0, 1);
            __builtin_prefetch(wb   + k0 + LDK, 0, 1);
        }
        const u16x8* a8 = (const u16x8*)(hrow + k0);
        *(u16x8*)dA       = a8[0];
        *(u16x8*)(dA + 8) = a8[1];
        stage16(wb + k0, dB);
        __syncthreads();
#pragma unroll
        for (int mi = 0; mi < 4; ++mi) {
            u16x16 af = ldsA_frag(lA, wm + mi * 16, lane);
#pragma unroll
            for (int ni = 0; ni < 2; ++ni) {
                u16x16 bf = ldsB_frag(lB, wn + ni * 16, lane);
                acc[mi][ni] = wmma_bf16(af, bf, acc[mi][ni]);
            }
        }
        __syncthreads();
    }

    const int cl = lane & 15;
    const int rh = (lane >> 4) * 8;
#pragma unroll
    for (int mi = 0; mi < 4; ++mi) {
#pragma unroll
        for (int ni = 0; ni < 2; ++ni) {
#pragma unroll
            for (int v = 0; v < 8; ++v) {
                int r = wm + mi * 16 + rh + v;
                if (r < rows_valid) {
                    out[(size_t)(row0 + r) * DMODEL + tile_n * TILE + wn + ni * 16 + cl] =
                        acc[mi][ni][v];
                }
            }
        }
    }
}

extern "C" void kernel_launch(void* const* d_in, const int* in_sizes, int n_in,
                              void* d_out, int out_size, void* d_ws, size_t ws_size,
                              hipStream_t stream) {
    const float* x    = (const float*)d_in[0];
    const float* wug  = (const float*)d_in[1];
    const float* wd   = (const float*)d_in[2];
    const int*   bse  = (const int*)d_in[3];
    float*       outp = (float*)d_out;
    unsigned short* h = (unsigned short*)d_ws;   // bf16 intermediate [n_tokens][HID]

    const int n_tokens = in_sizes[0] / DMODEL;   // 16384
    const int mtiles   = (n_tokens + TILE - 1) / TILE;

    dim3 blk(256);
    dim3 g1(HID / TILE,    mtiles);   // (8, 128)
    dim3 g2(DMODEL / TILE, mtiles);   // (16, 128)

    moe_upgate_silu_kernel<<<g1, blk, 0, stream>>>(x, wug, bse, h, n_tokens);
    moe_down_kernel<<<g2, blk, 0, stream>>>(h, wd, bse, outp, n_tokens);
}
